// CrossAttention_59871844106772
// MI455X (gfx1250) — matrix operand order repaired and checked by reading, not compiled, not run on hardware
//
#include <hip/hip_runtime.h>

// ---------------- types ----------------
typedef __attribute__((ext_vector_type(16))) __bf16 bf16x16;
typedef __attribute__((ext_vector_type(8)))  __bf16 bf16x8;
typedef __attribute__((ext_vector_type(8)))  float  f32x8;

#define EPSN 1e-6f

__device__ inline f32x8 zero8() {
    f32x8 z;
#pragma unroll
    for (int i = 0; i < 8; ++i) z[i] = 0.0f;
    return z;
}

// native RTNE f32 -> bf16 (v_cvt_pk_bf16_f32 on gfx1250)
__device__ inline __bf16 f2bf(float x) { return (__bf16)x; }

__device__ inline bf16x16 cat8(bf16x8 lo, bf16x8 hi) {
    return __builtin_shufflevector(lo, hi, 0, 1, 2, 3, 4, 5, 6, 7,
                                   8, 9, 10, 11, 12, 13, 14, 15);
}

__device__ inline f32x8 wmma_bf16(bf16x16 a, bf16x16 b, f32x8 c) {
    return __builtin_amdgcn_wmma_f32_16x16x32_bf16(
        false, a, false, b, (short)0, c, false, false);
}

// A-fragment (16x32, lane m = lane&15, half h = lane>>4):
// elem e : K = k0 + 16h + e
__device__ inline bf16x16 a_frag_bf(const __bf16* __restrict__ row, int k0, int h) {
    bf16x8 lo = *(const bf16x8*)(row + k0 + 16 * h);
    bf16x8 hi = *(const bf16x8*)(row + k0 + 16 * h + 8);
    return cat8(lo, hi);
}

// B-fragment (32x16, lane n = lane&15, half h = lane>>4):
// elem e : K = k0 + 16h + e; `col` points at K-contiguous data for column n
__device__ inline bf16x16 b_frag_bf(const __bf16* __restrict__ col, int k0, int h) {
    return *(const bf16x16*)(col + k0 + 16 * h);   // 32B contiguous -> 2x b128
}

// ---- CDNA5 async global->LDS copy (ASYNCcnt-tracked), 16B per lane ----
__device__ inline void async_ld16(unsigned lds_off, const __bf16* g) {
    asm volatile("global_load_async_to_lds_b128 %0, %1, off"
                 :: "v"(lds_off), "v"(g) : "memory");
}
template <int N>
__device__ inline void wait_async() {
    asm volatile("s_wait_asynccnt %0" :: "n"(N) : "memory");
}

// ---------------- kernel 0: fp32 -> bf16 repack (8 elems/thread) ----------------
__global__ __launch_bounds__(256) void cvt_bf16_kernel(
    const float* __restrict__ src, __bf16* __restrict__ dst, int n8)
{
    int i = blockIdx.x * 256 + threadIdx.x;
    if (i < n8) {
        f32x8 v = *(const f32x8*)(src + (size_t)i * 8);
        bf16x8 r;
#pragma unroll
        for (int e = 0; e < 8; ++e) r[e] = f2bf(v[e]);
        *(bf16x8*)(dst + (size_t)i * 8) = r;
    }
}

// ---------------- kernel 1: projections + normalize ----------------
// Each wave computes a 16x64 block = one head's D for 16 rows.
// mode 0: Xb=q_bf   Wb=Wq_bf  [C,C]   -> Qn[B,H,Lq,D]   (L2 norm, fold 1/tau)
// mode 1: Xb=kv_bf  Wb=Wkv_bf [2C,C]  -> Kn[B,H,Lkv,D] (norm) / Vt[B,H,D,Lkv]
__global__ __launch_bounds__(128) void proj_kernel(
    const __bf16* __restrict__ Xb, const __bf16* __restrict__ Wb,
    const float* __restrict__ tau_ptr, int mode,
    __bf16* __restrict__ Qn, __bf16* __restrict__ Kn, __bf16* __restrict__ Vt)
{
    constexpr int C = 1024, H = 16, D = 64, L = 1024;
    const int tid  = threadIdx.x;
    const int w    = tid >> 5;
    const int lane = tid & 31;
    const int n    = lane & 15;
    const int h    = lane >> 4;
    const int gr0  = blockIdx.x * 16;             // row block
    const int gc0  = blockIdx.y * 256 + w * 64;   // wave's 64-col chunk (one head)

    const __bf16* arow = Xb + (size_t)(gr0 + n) * C;
    const __bf16* wc0  = Wb + (size_t)(gc0 + 0 * 16 + n) * C;
    const __bf16* wc1  = Wb + (size_t)(gc0 + 1 * 16 + n) * C;
    const __bf16* wc2  = Wb + (size_t)(gc0 + 2 * 16 + n) * C;
    const __bf16* wc3  = Wb + (size_t)(gc0 + 3 * 16 + n) * C;

    f32x8 acc0 = zero8(), acc1 = zero8(), acc2 = zero8(), acc3 = zero8();
    for (int k0 = 0; k0 < C; k0 += 32) {
        bf16x16 af = a_frag_bf(arow, k0, h);           // shared by 4 tiles
        acc0 = wmma_bf16(af, b_frag_bf(wc0, k0, h), acc0);
        acc1 = wmma_bf16(af, b_frag_bf(wc1, k0, h), acc1);
        acc2 = wmma_bf16(af, b_frag_bf(wc2, k0, h), acc2);
        acc3 = wmma_bf16(af, b_frag_bf(wc3, k0, h), acc3);
    }

    const bool do_norm = (mode == 0) || (gc0 < 1024);
    float extra = 1.0f;
    if (mode == 0) extra = 1.0f / tau_ptr[0];

    const int sel = gc0 >> 10;                // kv: 0 = K, 1 = V
    const int hh  = (gc0 & 1023) >> 6;        // head index

#pragma unroll
    for (int e = 0; e < 8; ++e) {
        // wave-local row L2 norm: 4 tiles cover this head's 64 cols
        float ss = acc0[e] * acc0[e] + acc1[e] * acc1[e]
                 + acc2[e] * acc2[e] + acc3[e] * acc3[e];
        ss += __shfl_xor(ss, 1);
        ss += __shfl_xor(ss, 2);
        ss += __shfl_xor(ss, 4);
        ss += __shfl_xor(ss, 8);
        float sc = do_norm ? (extra / (sqrtf(ss) + EPSN)) : 1.0f;

        int grow = gr0 + e + 8 * h;
        int bb = grow >> 10;
        int l  = grow & 1023;
        size_t bhl = ((size_t)(bb * H + hh) * L + l);
        if (mode == 0) {
            __bf16* dst = Qn + (bhl << 6);
            dst[0 * 16 + n] = f2bf(acc0[e] * sc);
            dst[1 * 16 + n] = f2bf(acc1[e] * sc);
            dst[2 * 16 + n] = f2bf(acc2[e] * sc);
            dst[3 * 16 + n] = f2bf(acc3[e] * sc);
        } else if (sel == 0) {
            __bf16* dst = Kn + (bhl << 6);
            dst[0 * 16 + n] = f2bf(acc0[e] * sc);
            dst[1 * 16 + n] = f2bf(acc1[e] * sc);
            dst[2 * 16 + n] = f2bf(acc2[e] * sc);
            dst[3 * 16 + n] = f2bf(acc3[e] * sc);
        } else {
            __bf16* dst = Vt + ((size_t)(bb * H + hh) * D) * L + l;
            dst[(size_t)(0 * 16 + n) * L] = f2bf(acc0[e]);
            dst[(size_t)(1 * 16 + n) * L] = f2bf(acc1[e]);
            dst[(size_t)(2 * 16 + n) * L] = f2bf(acc2[e]);
            dst[(size_t)(3 * 16 + n) * L] = f2bf(acc3[e]);
        }
    }
}

// ---------------- kernel 2: flash attention ----------------
// 4 waves/block share one (batch, head); K/V slabs double-buffered in LDS via
// global_load_async_to_lds_b128; each wave owns one 16-row Q block.
__global__ __launch_bounds__(128) void attn_kernel(
    const __bf16* __restrict__ Qn, const __bf16* __restrict__ Kn,
    const __bf16* __restrict__ Vt, __bf16* __restrict__ AO)
{
    constexpr int D = 64, L = 1024, C = 1024;
    __shared__ __align__(16) __bf16 Ks[2][32 * 64];   // [buf][kv][d]
    __shared__ __align__(16) __bf16 Vs[2][64 * 32];   // [buf][d][kv]
    __shared__ __align__(16) __bf16 Pld[4][16 * 32];  // per-wave P transpose tile

    const int tid  = threadIdx.x;
    const int w    = tid >> 5;
    const int lane = tid & 31;
    const int n    = lane & 15;
    const int h    = lane >> 4;

    const int bh = blockIdx.x >> 4;             // b*H + head  (0..127)
    const int qb = (blockIdx.x & 15) * 4 + w;   // 16-row q block (0..63)

    const __bf16* qp = Qn + ((size_t)bh * L + qb * 16) * D;
    const __bf16* kp = Kn + (size_t)bh * L * D;
    const __bf16* vp = Vt + (size_t)bh * D * L;

    bf16x16 aq0 = a_frag_bf(qp + n * D, 0, h);
    bf16x16 aq1 = a_frag_bf(qp + n * D, 32, h);

    f32x8 acc0 = zero8(), acc1 = zero8(), acc2 = zero8(), acc3 = zero8();
    float mrow[8], lrow[8];
#pragma unroll
    for (int e = 0; e < 8; ++e) { mrow[e] = -1e30f; lrow[e] = 0.0f; }

    __bf16* pw = &Pld[w][0];

    // prologue: stage first slab (each thread: 2 K chunks + 2 V chunks, 16B each)
    {
        unsigned ksb = (unsigned)(unsigned long long)&Ks[0][0];
        unsigned vsb = (unsigned)(unsigned long long)&Vs[0][0];
#pragma unroll
        for (int j = 0; j < 2; ++j) {
            int c = tid + j * 128;
            async_ld16(ksb + c * 16, kp + (size_t)(c >> 3) * 64 + (c & 7) * 8);
            async_ld16(vsb + c * 16, vp + (size_t)(c >> 2) * 1024 + (c & 3) * 8);
        }
    }

    for (int kv0 = 0; kv0 < L; kv0 += 32) {
        const int cur = (kv0 >> 5) & 1;
        const bool more = (kv0 + 32) < L;
        if (more) {
            unsigned ksb = (unsigned)(unsigned long long)&Ks[cur ^ 1][0];
            unsigned vsb = (unsigned)(unsigned long long)&Vs[cur ^ 1][0];
#pragma unroll
            for (int j = 0; j < 2; ++j) {
                int c = tid + j * 128;
                async_ld16(ksb + c * 16, kp + (size_t)(kv0 + 32 + (c >> 3)) * 64 + (c & 7) * 8);
                async_ld16(vsb + c * 16, vp + (size_t)(c >> 2) * 1024 + kv0 + 32 + (c & 3) * 8);
            }
            wait_async<4>();      // current slab resident; next slab in flight
        } else {
            wait_async<0>();
        }
        __syncthreads();

        const __bf16* ks = &Ks[cur][0];
        const __bf16* vs = &Vs[cur][0];

        // S = Qn * Kn^T over this 32-wide KV slab (two 16-col tiles)
        f32x8 s0 = zero8(), s1 = zero8();
        s0 = wmma_bf16(aq0, b_frag_bf(ks + (size_t)n * 64, 0, h), s0);
        s0 = wmma_bf16(aq1, b_frag_bf(ks + (size_t)n * 64, 32, h), s0);
        s1 = wmma_bf16(aq0, b_frag_bf(ks + (size_t)(16 + n) * 64, 0, h), s1);
        s1 = wmma_bf16(aq1, b_frag_bf(ks + (size_t)(16 + n) * 64, 32, h), s1);

        // online softmax (row stats per C-layout element, reduced over 16 lanes)
#pragma unroll
        for (int e = 0; e < 8; ++e) {
            float t = fmaxf(s0[e], s1[e]);
            t = fmaxf(t, __shfl_xor(t, 1));
            t = fmaxf(t, __shfl_xor(t, 2));
            t = fmaxf(t, __shfl_xor(t, 4));
            t = fmaxf(t, __shfl_xor(t, 8));
            float mn   = fmaxf(mrow[e], t);
            float corr = __expf(mrow[e] - mn);
            float p0   = __expf(s0[e] - mn);
            float p1   = __expf(s1[e] - mn);
            float ts = p0 + p1;
            ts += __shfl_xor(ts, 1);
            ts += __shfl_xor(ts, 2);
            ts += __shfl_xor(ts, 4);
            ts += __shfl_xor(ts, 8);
            lrow[e] = lrow[e] * corr + ts;
            mrow[e] = mn;
            acc0[e] *= corr; acc1[e] *= corr; acc2[e] *= corr; acc3[e] *= corr;
            pw[(e + 8 * h) * 32 + n]      = f2bf(p0);
            pw[(e + 8 * h) * 32 + 16 + n] = f2bf(p1);
        }
        __syncthreads();
        // re-read P as an A-fragment (C-layout -> A-layout transpose via LDS)
        bf16x16 ap = a_frag_bf(pw + n * 32, 0, h);

        // acc += P * V  (4 n-subtiles of 16 over D=64)
        acc0 = wmma_bf16(ap, b_frag_bf(vs + (size_t)(0 * 16 + n) * 32, 0, h), acc0);
        acc1 = wmma_bf16(ap, b_frag_bf(vs + (size_t)(1 * 16 + n) * 32, 0, h), acc1);
        acc2 = wmma_bf16(ap, b_frag_bf(vs + (size_t)(2 * 16 + n) * 32, 0, h), acc2);
        acc3 = wmma_bf16(ap, b_frag_bf(vs + (size_t)(3 * 16 + n) * 32, 0, h), acc3);
        __syncthreads();   // protect cur slab + Pld before they are rewritten
    }

    // epilogue: divide by softmax denominator, store bf16 to AO[B,Lq,C]
    const int b  = bh >> 4;
    const int hh = bh & 15;
#pragma unroll
    for (int e = 0; e < 8; ++e) {
        float inv = 1.0f / lrow[e];
        int m = e + 8 * h;
        __bf16* orow = AO + ((size_t)(b * L + qb * 16 + m)) * C + hh * D;
        orow[0 * 16 + n] = f2bf(acc0[e] * inv);
        orow[1 * 16 + n] = f2bf(acc1[e] * inv);
        orow[2 * 16 + n] = f2bf(acc2[e] * inv);
        orow[3 * 16 + n] = f2bf(acc3[e] * inv);
    }
}

// ---------------- kernel 3: output projection + bias (fp32 out) ----------------
__global__ __launch_bounds__(128) void outproj_kernel(
    const __bf16* __restrict__ AO, const __bf16* __restrict__ Wpb,
    const float* __restrict__ bias, float* __restrict__ out)
{
    constexpr int C = 1024;
    const int tid  = threadIdx.x;
    const int w    = tid >> 5;
    const int lane = tid & 31;
    const int n    = lane & 15;
    const int h    = lane >> 4;
    const int gr0  = blockIdx.x * 16;
    const int gc0  = blockIdx.y * 256 + w * 64;

    const __bf16* arow = AO + (size_t)(gr0 + n) * C;
    const __bf16* wc0  = Wpb + (size_t)(gc0 + 0 * 16 + n) * C;
    const __bf16* wc1  = Wpb + (size_t)(gc0 + 1 * 16 + n) * C;
    const __bf16* wc2  = Wpb + (size_t)(gc0 + 2 * 16 + n) * C;
    const __bf16* wc3  = Wpb + (size_t)(gc0 + 3 * 16 + n) * C;

    f32x8 acc0 = zero8(), acc1 = zero8(), acc2 = zero8(), acc3 = zero8();
    for (int k0 = 0; k0 < C; k0 += 32) {
        bf16x16 af = a_frag_bf(arow, k0, h);
        acc0 = wmma_bf16(af, b_frag_bf(wc0, k0, h), acc0);
        acc1 = wmma_bf16(af, b_frag_bf(wc1, k0, h), acc1);
        acc2 = wmma_bf16(af, b_frag_bf(wc2, k0, h), acc2);
        acc3 = wmma_bf16(af, b_frag_bf(wc3, k0, h), acc3);
    }

    float b0 = bias[gc0 + 0 * 16 + n];
    float b1 = bias[gc0 + 1 * 16 + n];
    float b2 = bias[gc0 + 2 * 16 + n];
    float b3 = bias[gc0 + 3 * 16 + n];
#pragma unroll
    for (int e = 0; e < 8; ++e) {
        int m = e + 8 * h;
        float* orow = out + (size_t)(gr0 + m) * C + gc0;
        orow[0 * 16 + n] = acc0[e] + b0;
        orow[1 * 16 + n] = acc1[e] + b1;
        orow[2 * 16 + n] = acc2[e] + b2;
        orow[3 * 16 + n] = acc3[e] + b3;
    }
}

// ---------------- host launcher ----------------
extern "C" void kernel_launch(void* const* d_in, const int* in_sizes, int n_in,
                              void* d_out, int out_size, void* d_ws, size_t ws_size,
                              hipStream_t stream) {
    (void)in_sizes; (void)n_in; (void)out_size; (void)ws_size;
    const float* q     = (const float*)d_in[0];
    const float* kv    = (const float*)d_in[1];
    const float* Wq    = (const float*)d_in[2];
    const float* Wkv   = (const float*)d_in[3];
    const float* Wproj = (const float*)d_in[4];
    const float* bproj = (const float*)d_in[5];
    const float* tau   = (const float*)d_in[6];
    float* out = (float*)d_out;

    char* ws = (char*)d_ws;
    __bf16* Qn   = (__bf16*)(ws + ((size_t)0   << 20));  // 16 MB [B,H,Lq,D]
    __bf16* Kn   = (__bf16*)(ws + ((size_t)16  << 20));  // 16 MB [B,H,Lkv,D]
    __bf16* Vt   = (__bf16*)(ws + ((size_t)32  << 20));  // 16 MB [B,H,D,Lkv]
    __bf16* AO   = (__bf16*)(ws + ((size_t)48  << 20));  // 16 MB [B,Lq,C]
    __bf16* qb   = (__bf16*)(ws + ((size_t)64  << 20));  // 16 MB bf16(q)
    __bf16* kvb  = (__bf16*)(ws + ((size_t)80  << 20));  // 16 MB bf16(kv)
    __bf16* Wqb  = (__bf16*)(ws + ((size_t)96  << 20));  //  2 MB bf16(Wq)
    __bf16* Wkvb = (__bf16*)(ws + ((size_t)98  << 20));  //  4 MB bf16(Wkv)
    __bf16* Wpb  = (__bf16*)(ws + ((size_t)102 << 20));  //  2 MB bf16(Wproj)

    dim3 blk(128);
    // repack to bf16 (8 elems / thread)
    cvt_bf16_kernel<<<4096, 256, 0, stream>>>(q,     qb,   8 * 1024 * 1024 / 8);
    cvt_bf16_kernel<<<4096, 256, 0, stream>>>(kv,    kvb,  8 * 1024 * 1024 / 8);
    cvt_bf16_kernel<<<512,  256, 0, stream>>>(Wq,    Wqb,  1024 * 1024 / 8);
    cvt_bf16_kernel<<<1024, 256, 0, stream>>>(Wkv,   Wkvb, 2 * 1024 * 1024 / 8);
    cvt_bf16_kernel<<<512,  256, 0, stream>>>(Wproj, Wpb,  1024 * 1024 / 8);

    // projections (each wave: 16x64 = one head chunk)
    proj_kernel<<<dim3(512, 4), blk, 0, stream>>>(qb,  Wqb,  tau, 0, Qn, Kn, Vt);
    proj_kernel<<<dim3(512, 8), blk, 0, stream>>>(kvb, Wkvb, tau, 1, Qn, Kn, Vt);
    // attention
    attn_kernel<<<dim3(2048), blk, 0, stream>>>(Qn, Kn, Vt, AO);
    // output projection
    outproj_kernel<<<dim3(512, 4), blk, 0, stream>>>(AO, Wpb, bproj, out);
}
